// MultiHeadAttention_32478542693167
// MI455X (gfx1250) — compile-verified, matrix operand order checked by reading, not run on hardware
//
#include <hip/hip_runtime.h>
#include <hip/hip_bf16.h>

// ---------------------------------------------------------------------------
// MHA forward for gfx1250 (MI455X). f16 WMMA (16x16x32) with fp32 accumulate.
// GEMMs use 32x64 register tiles per wave (halves B/L2 traffic vs 16x64).
// ---------------------------------------------------------------------------

typedef __attribute__((ext_vector_type(16))) _Float16 v16h;
typedef __attribute__((ext_vector_type(8)))  _Float16 v8h;
typedef __attribute__((ext_vector_type(8)))  float    v8f;

constexpr int Bc  = 4;
constexpr int Sc  = 1024;
constexpr int Dc  = 1024;
constexpr int Hc  = 16;
constexpr int DKc = 64;

__device__ inline v16h cat8(v8h x, v8h y) {
  return __builtin_shufflevector(x, y, 0,1,2,3,4,5,6,7,8,9,10,11,12,13,14,15);
}

// A/B operand loader: per-lane 16 f16 with K pattern {ka+0..7, ka+16..23}.
// (caller passes p already offset by ka = k0 + half*8)
__device__ inline v16h load_ab(const _Float16* p) {
  const v8h* vp = (const v8h*)p;
  return cat8(vp[0], vp[2]);          // +0..7 and +16..23 (f16 units)
}
__device__ inline v16h load_ab(const float* p) {   // fused fp32 -> f16 convert
  v16h a;
#pragma unroll
  for (int i = 0; i < 8; ++i) { a[i] = (_Float16)p[i]; a[i + 8] = (_Float16)p[i + 16]; }
  return a;
}

__device__ inline v8f vzero8() {
  v8f z = {0.f, 0.f, 0.f, 0.f, 0.f, 0.f, 0.f, 0.f};
  return z;
}

// ---------------------------------------------------------------------------
// W [K][N] fp32  ->  WT [N][K] f16
// ---------------------------------------------------------------------------
__global__ __launch_bounds__(256) void convert_transpose(
    const float* __restrict__ W, _Float16* __restrict__ WT, int K, int N) {
  int idx = blockIdx.x * blockDim.x + threadIdx.x;
  if (idx >= K * N) return;
  int k = idx / N;
  int n = idx - k * N;
  WT[(size_t)n * K + k] = (_Float16)W[idx];
}

// ---------------------------------------------------------------------------
// C[M][N] = A[M][K] * BT[N][K]^T + bias.  One wave computes a 32x64 tile
// (B regs reused across 2 M-tiles, A regs reused across 4 N-tiles).
// A is fp32 (converted on load) or f16.
// ---------------------------------------------------------------------------
template <typename AT>
__global__ __launch_bounds__(256) void wmma_gemm(
    const AT* __restrict__ A, const _Float16* __restrict__ BT,
    const float* __restrict__ bias, float* __restrict__ C,
    int M, int N, int K) {
  const int lane = threadIdx.x & 31;
  const int wave = (int)((blockIdx.x * blockDim.x + threadIdx.x) >> 5);
  const int ntg  = N >> 6;                 // 64-wide col groups
  const int tm   = wave / ntg;             // 32-row tile index
  const int ng   = wave - tm * ntg;
  if (tm * 32 >= M) return;
  const int l    = lane & 15;
  const int half = lane >> 4;
  const size_t arow0 = (size_t)(tm * 32 + l);
  const size_t arow1 = arow0 + 16;

  v8f acc[8] = {vzero8(), vzero8(), vzero8(), vzero8(),
                vzero8(), vzero8(), vzero8(), vzero8()};

  for (int k0 = 0; k0 < K; k0 += 32) {
    const int ka = k0 + half * 8;
    v16h a0 = load_ab(A + arow0 * (size_t)K + ka);
    v16h a1 = load_ab(A + arow1 * (size_t)K + ka);
    const _Float16* bbase = BT + (size_t)(ng * 64 + l) * K + ka;
#pragma unroll
    for (int t = 0; t < 4; ++t) {
      v16h bm = load_ab(bbase + (size_t)(16 * t) * K);
      acc[t] = __builtin_amdgcn_wmma_f32_16x16x32_f16(
          false, a0, false, bm, (short)0, acc[t], false, false);
      acc[4 + t] = __builtin_amdgcn_wmma_f32_16x16x32_f16(
          false, a1, false, bm, (short)0, acc[4 + t], false, false);
    }
  }

  // C layout: VGPR r -> row r + half*8 within each 16-row tile
#pragma unroll
  for (int g = 0; g < 2; ++g) {
    const int rbase = tm * 32 + g * 16 + half * 8;
#pragma unroll
    for (int t = 0; t < 4; ++t) {
      const int col = ng * 64 + t * 16 + l;
      const float bv = bias[col];
#pragma unroll
      for (int r = 0; r < 8; ++r)
        C[(size_t)(rbase + r) * N + col] = acc[4 * g + t][r] + bv;
    }
  }
}

// ---------------------------------------------------------------------------
// RoPE on qw/kw, pack to f16:
//   Qh,Kh: [B,H,S,DK]   (row-major over d  -> A / "BT" layout for Q·K^T)
//   VT:    [B,H,DK,S]   (row-major over s  -> "BT" layout for P·V)
// ---------------------------------------------------------------------------
__global__ __launch_bounds__(256) void rope_pack(
    const float* __restrict__ qw, const float* __restrict__ kw,
    const float* __restrict__ vw, const float* __restrict__ sp,
    _Float16* __restrict__ Qh, _Float16* __restrict__ Kh,
    _Float16* __restrict__ VT) {
  const int P = DKc / 2;                     // 32 pairs
  int idx = blockIdx.x * blockDim.x + threadIdx.x;
  int p = idx & (P - 1);
  int h = (idx / P) & (Hc - 1);
  int s = (idx / (P * Hc)) & (Sc - 1);
  int b = idx / (P * Hc * Sc);
  if (b >= Bc) return;

  size_t src = (((size_t)b * Sc + s) * Hc + h) * DKc + 2 * p;
  float sn = sp[(size_t)s * DKc + 2 * p];
  float cs = sp[(size_t)s * DKc + 2 * p + 1];

  float q0 = qw[src], q1 = qw[src + 1];
  float k0 = kw[src], k1 = kw[src + 1];
  float v0 = vw[src], v1 = vw[src + 1];

  size_t dst = (((size_t)b * Hc + h) * Sc + s) * DKc + 2 * p;
  Qh[dst]     = (_Float16)(q0 * cs - q1 * sn);
  Qh[dst + 1] = (_Float16)(q1 * cs + q0 * sn);
  Kh[dst]     = (_Float16)(k0 * cs - k1 * sn);
  Kh[dst + 1] = (_Float16)(k1 * cs + k0 * sn);

  size_t vt = (((size_t)b * Hc + h) * DKc + 2 * p) * Sc + s;
  VT[vt]      = (_Float16)v0;
  VT[vt + Sc] = (_Float16)v1;
}

// ---------------------------------------------------------------------------
// Flash attention: one wave per (b, h, 16 query rows). Keys streamed in
// chunks of 32 with online softmax. ctx: f16 [B,S,H*DK].
// ---------------------------------------------------------------------------
__global__ __launch_bounds__(256) void flash_attn(
    const _Float16* __restrict__ Qh, const _Float16* __restrict__ Kh,
    const _Float16* __restrict__ VT, const int* __restrict__ vmask,
    _Float16* __restrict__ ctx) {
  const int lane = threadIdx.x & 31;
  const int wv   = threadIdx.x >> 5;                     // wave in block
  const int wave = (int)((blockIdx.x * blockDim.x + threadIdx.x) >> 5);
  const int qtiles = Sc / 16;
  const int qt = wave % qtiles;
  const int h  = (wave / qtiles) % Hc;
  const int b  = wave / (qtiles * Hc);
  const int l    = lane & 15;
  const int half = lane >> 4;

  __shared__ _Float16 pbuf[8][16 * 32];                  // 1KB per wave
  _Float16* Pl = pbuf[wv];

  const _Float16* Qrow  = Qh + (((size_t)b * Hc + h) * Sc + qt * 16 + l) * DKc;
  const _Float16* Kbase = Kh + (((size_t)b * Hc + h) * Sc) * DKc;
  const _Float16* Vbase = VT + (((size_t)b * Hc + h) * DKc) * Sc;

  float m[8], lsum[8];
  v8f o[4] = {vzero8(), vzero8(), vzero8(), vzero8()};
#pragma unroll
  for (int r = 0; r < 8; ++r) { m[r] = -1e30f; lsum[r] = 0.f; }

  const float scale = 0.125f;                            // 1/sqrt(64)

  for (int kc = 0; kc < Sc; kc += 32) {
    // ---- scores: 16 q-rows x 32 keys = two 16x16 C tiles ----
    v8f s0 = vzero8(), s1 = vzero8();
#pragma unroll
    for (int kk = 0; kk < DKc; kk += 32) {
      const int ka = kk + half * 8;
      v16h a  = load_ab(Qrow + ka);
      v16h b0 = load_ab(Kbase + (size_t)(kc + l) * DKc + ka);
      v16h b1 = load_ab(Kbase + (size_t)(kc + 16 + l) * DKc + ka);
      s0 = __builtin_amdgcn_wmma_f32_16x16x32_f16(false, a, false, b0, (short)0, s0, false, false);
      s1 = __builtin_amdgcn_wmma_f32_16x16x32_f16(false, a, false, b1, (short)0, s1, false, false);
    }

    // ---- mask + scale (column = key index = kc + l / kc + 16 + l) ----
    const int ok0 = vmask[(size_t)b * Sc + kc + l];
    const int ok1 = vmask[(size_t)b * Sc + kc + 16 + l];
#pragma unroll
    for (int r = 0; r < 8; ++r) {
      s0[r] = ok0 ? s0[r] * scale : -1e30f;
      s1[r] = ok1 ? s1[r] * scale : -1e30f;
    }

    // ---- online softmax per row (row r + half*8); reduce across 16 lanes --
#pragma unroll
    for (int r = 0; r < 8; ++r) {
      float rm = fmaxf(s0[r], s1[r]);
#pragma unroll
      for (int msk = 1; msk < 16; msk <<= 1) rm = fmaxf(rm, __shfl_xor(rm, msk, 16));
      float mn = fmaxf(m[r], rm);
      float p0 = __expf(s0[r] - mn);
      float p1 = __expf(s1[r] - mn);
      float rs = p0 + p1;
#pragma unroll
      for (int msk = 1; msk < 16; msk <<= 1) rs += __shfl_xor(rs, msk, 16);
      float corr = __expf(m[r] - mn);
      lsum[r] = lsum[r] * corr + rs;
      m[r] = mn;
      o[0][r] *= corr; o[1][r] *= corr; o[2][r] *= corr; o[3][r] *= corr;
      Pl[(r + half * 8) * 32 + l]      = (_Float16)p0;
      Pl[(r + half * 8) * 32 + 16 + l] = (_Float16)p1;
    }
    // same-wave DS ops are in-order; compiler inserts s_wait_dscnt for the RAW.

    // ---- O += P(16x32) @ V(32x64): re-load P in A layout from LDS ----
    v16h pa = load_ab(Pl + l * 32 + half * 8);
#pragma unroll
    for (int t = 0; t < 4; ++t) {
      const _Float16* vb = Vbase + (size_t)(t * 16 + l) * Sc + kc + half * 8;
      v16h bm = load_ab(vb);
      o[t] = __builtin_amdgcn_wmma_f32_16x16x32_f16(false, pa, false, bm, (short)0, o[t], false, false);
    }
  }

  // ---- normalize + store ctx[b][s][h*64 + d] as f16 ----
  const size_t srow = (size_t)qt * 16 + half * 8;
#pragma unroll
  for (int t = 0; t < 4; ++t) {
#pragma unroll
    for (int r = 0; r < 8; ++r) {
      float val = o[t][r] / lsum[r];
      ctx[(((size_t)b * Sc + srow + r) * (Hc * DKc)) + h * DKc + t * 16 + l] =
          (_Float16)val;
    }
  }
}

// ---------------------------------------------------------------------------
// Host launch
// ---------------------------------------------------------------------------
extern "C" void kernel_launch(void* const* d_in, const int* in_sizes, int n_in,
                              void* d_out, int out_size, void* d_ws, size_t ws_size,
                              hipStream_t stream) {
  (void)in_sizes; (void)n_in; (void)out_size; (void)ws_size;

  const float* q  = (const float*)d_in[0];
  const float* k  = (const float*)d_in[1];
  const float* v  = (const float*)d_in[2];
  const float* sp = (const float*)d_in[3];
  const int*   vm = (const int*)d_in[4];
  const float* Wq = (const float*)d_in[5];
  const float* bq = (const float*)d_in[6];
  const float* Wk = (const float*)d_in[7];
  const float* bk = (const float*)d_in[8];
  const float* Wv = (const float*)d_in[9];
  const float* bv = (const float*)d_in[10];
  const float* Wo = (const float*)d_in[11];
  const float* bo = (const float*)d_in[12];
  float* out = (float*)d_out;

  char* ws = (char*)d_ws;
  const size_t MB = 1ull << 20;
  _Float16* WTq = (_Float16*)(ws + 0 * MB);    // 2 MB each (1024x1024 f16)
  _Float16* WTk = (_Float16*)(ws + 2 * MB);
  _Float16* WTv = (_Float16*)(ws + 4 * MB);
  _Float16* WTo = (_Float16*)(ws + 6 * MB);
  float*    qw  = (float*)(ws + 8 * MB);       // 16 MB each (4096x1024 f32)
  float*    kw  = (float*)(ws + 24 * MB);
  float*    vw  = (float*)(ws + 40 * MB);
  _Float16* QhP = (_Float16*)(ws + 56 * MB);   // 8 MB each
  _Float16* KhP = (_Float16*)(ws + 64 * MB);
  _Float16* VTP = (_Float16*)(ws + 72 * MB);
  _Float16* ctx = (_Float16*)(ws + 80 * MB);   // 8 MB   -> 88 MB total

  const int HD = Hc * DKc;                     // 1024
  const int nW = Dc * HD;                      // 1M elements per weight
  convert_transpose<<<nW / 256, 256, 0, stream>>>(Wq, WTq, Dc, HD);
  convert_transpose<<<nW / 256, 256, 0, stream>>>(Wk, WTk, Dc, HD);
  convert_transpose<<<nW / 256, 256, 0, stream>>>(Wv, WTv, Dc, HD);
  convert_transpose<<<nW / 256, 256, 0, stream>>>(Wo, WTo, HD, Dc);

  const int M = Bc * Sc;                       // 4096
  const int gwaves  = (M / 32) * (HD / 64);    // 2048 waves (32x64 tiles)
  const int gblocks = gwaves * 32 / 256;       // 256 blocks
  wmma_gemm<float><<<gblocks, 256, 0, stream>>>(q, WTq, bq, qw, M, HD, Dc);
  wmma_gemm<float><<<gblocks, 256, 0, stream>>>(k, WTk, bk, kw, M, HD, Dc);
  wmma_gemm<float><<<gblocks, 256, 0, stream>>>(v, WTv, bv, vw, M, HD, Dc);

  const int rthreads = Bc * Sc * Hc * (DKc / 2);   // 2,097,152
  rope_pack<<<rthreads / 256, 256, 0, stream>>>(qw, kw, vw, sp, QhP, KhP, VTP);

  const int awaves = Bc * Hc * (Sc / 16);          // 4096 waves
  flash_attn<<<awaves * 32 / 256, 256, 0, stream>>>(QhP, KhP, VTP, vm, ctx);

  wmma_gemm<_Float16><<<gblocks, 256, 0, stream>>>(ctx, WTo, bo, out, M, Dc, HD);
}
